// RNN_arch_2_65670049955962
// MI455X (gfx1250) — matrix operand order repaired and checked by reading, not compiled, not run on hardware
//
#include <hip/hip_runtime.h>
#include <cmath>

typedef _Float16 f16;
typedef __attribute__((ext_vector_type(16))) _Float16 v16h;
typedef __attribute__((ext_vector_type(8)))  _Float16 v8h;
typedef __attribute__((ext_vector_type(8)))  float    v8f;
typedef __attribute__((ext_vector_type(4)))  float    v4f;

#define T_STEPS 16
#define BATCH   16384
#define D_IN    64
#define D_H     256
#define D_MID   64
#define D_OUT   4

// ---- packed f16 weight layout in d_ws (B-operand order) ----
// Each 32x16 (KxN) tile = 512 f16: lane = n + 16*kh, lane holds W[j0+n, k0+kh*8 + {0..7, 16..23}]
#define PK_I2H_OFF 0
#define PK_I2H_N   (16*2*512)                 // W_i2h: 16 n-tiles x 2 k-tiles
#define PK_H2H_OFF (PK_I2H_OFF + PK_I2H_N)
#define PK_H2H_N   (16*8*512)                 // W_h2h: 16 x 8
#define PK_H2O_OFF (PK_H2H_OFF + PK_H2H_N)
#define PK_H2O_N   (4*8*512)                  // W_h2o: 4 x 8
#define PK_FC_OFF  (PK_H2O_OFF + PK_H2O_N)
#define PK_FC_N    (1*2*512)                  // W_fc padded to 16 rows: 1 x 2
#define PK_TOTAL   (PK_FC_OFF + PK_FC_N)      // 99328 f16 = 198656 bytes

// ---- LDS layout ----
#define SM_W_BYTES   (PK_TOTAL*2)             // 198656
#define SM_BSUM_OFF  SM_W_BYTES               // 256 f32
#define SM_BH2O_OFF  (SM_BSUM_OFF + 256*4)    // 64 f32
#define SM_HSCR_OFF  (SM_BH2O_OFF + 256)      // 199936
#define H_PITCH      264                      // f16 per scratch row (pad vs 256)
#define SM_HSCR_WAVE (16*H_PITCH*2)           // 8448 B per wave
#define NWAVES       4
#define SM_TOTAL     (SM_HSCR_OFF + NWAVES*SM_HSCR_WAVE)  // 233728 B

__device__ __forceinline__ float fast_tanh(float v) {
#if __has_builtin(__builtin_amdgcn_tanhf)
  return __builtin_amdgcn_tanhf(v);
#elif __has_builtin(__builtin_amdgcn_tanh_f32)
  return __builtin_amdgcn_tanh_f32(v);
#else
  return tanhf(v);
#endif
}

__device__ __forceinline__ v16h combine8(v8h lo, v8h hi) {
  v16h r;
#pragma unroll
  for (int i = 0; i < 8; i++) { r[i] = lo[i]; r[i + 8] = hi[i]; }
  return r;
}

// A-operand (16x32 f16) from LDS scratch: rowbase = scratch + (lane&15)*H_PITCH
// lane<16: K = kt*32 + {0..7, 16..23}; lane>=16: K = kt*32 + {8..15, 24..31}
__device__ __forceinline__ v16h load_A_lds(const f16* rowbase, int kt, int hsel) {
  const f16* p = rowbase + kt * 32 + hsel * 8;
  v8h lo = *(const v8h*)p;
  v8h hi = *(const v8h*)(p + 16);
  return combine8(lo, hi);
}

// A-operand from a global f32 row (x or hc1), converting to f16
__device__ __forceinline__ v16h load_A_gf32(const float* rowptr, int kt, int hsel) {
  const float* p = rowptr + kt * 32 + hsel * 8;
  v4f a0 = *(const v4f*)p,        a1 = *(const v4f*)(p + 4);
  v4f b0 = *(const v4f*)(p + 16), b1 = *(const v4f*)(p + 20);
  v16h r;
#pragma unroll
  for (int i = 0; i < 4; i++) {
    r[i]      = (_Float16)a0[i];
    r[i + 4]  = (_Float16)a1[i];
    r[i + 8]  = (_Float16)b0[i];
    r[i + 12] = (_Float16)b1[i];
  }
  return r;
}

// B-operand (32x16 f16) from packed LDS array; blk = nt*KT + kt
__device__ __forceinline__ v16h load_B_lds(const f16* arr, int blk, int lane) {
  const f16* p = arr + blk * 512 + lane * 16;
  v8h lo = *(const v8h*)p;
  v8h hi = *(const v8h*)(p + 8);
  return combine8(lo, hi);
}

#define WMMA(a, b, c) \
  __builtin_amdgcn_wmma_f32_16x16x32_f16(false, (a), false, (b), (short)0, (c), false, false)

// ---------------- weight packing kernel ----------------
__global__ void pack_weights(const float* __restrict__ Wi2h, const float* __restrict__ Wh2h,
                             const float* __restrict__ Wh2o, const float* __restrict__ Wfc,
                             f16* __restrict__ ws) {
  int p = blockIdx.x * blockDim.x + threadIdx.x;
  if (p >= PK_TOTAL) return;
  const float* W; int J, K, KT, base;
  if (p < PK_H2H_OFF)      { W = Wi2h; J = 256; K = 64;  KT = 2; base = PK_I2H_OFF; }
  else if (p < PK_H2O_OFF) { W = Wh2h; J = 256; K = 256; KT = 8; base = PK_H2H_OFF; }
  else if (p < PK_FC_OFF)  { W = Wh2o; J = 64;  K = 256; KT = 8; base = PK_H2O_OFF; }
  else                     { W = Wfc;  J = 4;   K = 64;  KT = 2; base = PK_FC_OFF; }
  int q  = p - base;
  int blk = q >> 9, r = q & 511, ln = r >> 4, e = r & 15;
  int nt = blk / KT, kt = blk % KT;
  int j = nt * 16 + (ln & 15);
  int k = kt * 32 + (ln >> 4) * 8 + (e & 7) + (e >> 3) * 16;
  float v = (j < J) ? W[j * K + k] : 0.0f;
  ws[p] = (f16)v;
}

// ---------------- main RNN kernel ----------------
__global__ void __launch_bounds__(128)
rnn_wmma_kernel(const float* __restrict__ x, const float* __restrict__ hc1,
                const float* __restrict__ b_i2h, const float* __restrict__ b_h2h,
                const float* __restrict__ b_h2o, const float* __restrict__ b_fc,
                const f16* __restrict__ wpack,
                float* __restrict__ out_seq, float* __restrict__ h_final) {
  extern __shared__ char smem[];
  f16*   sW    = (f16*)smem;
  const f16* sWi2h = sW + PK_I2H_OFF;
  const f16* sWh2h = sW + PK_H2H_OFF;
  const f16* sWh2o = sW + PK_H2O_OFF;
  const f16* sWfc  = sW + PK_FC_OFF;
  float* s_bsum = (float*)(smem + SM_BSUM_OFF);
  float* s_bh2o = (float*)(smem + SM_BH2O_OFF);

  const int tid = threadIdx.x;
  // stage packed weights into LDS (global b128 -> ds b128)
  for (int off = tid * 16; off < SM_W_BYTES; off += 128 * 16)
    *(int4*)(smem + off) = *(const int4*)((const char*)wpack + off);
  s_bsum[tid]       = b_i2h[tid]       + b_h2h[tid];
  s_bsum[tid + 128] = b_i2h[tid + 128] + b_h2h[tid + 128];
  if (tid < 64) s_bh2o[tid] = b_h2o[tid];
  __syncthreads();

  const int lane = tid & 31, w = tid >> 5;
  const int lanelo = lane & 15, hsel = lane >> 4;
  const int brow0 = blockIdx.x * (NWAVES * 16) + w * 16;  // wave's batch base
  const int rowg  = brow0 + lanelo;                        // A-operand row (global)
  f16* hscr = (f16*)(smem + SM_HSCR_OFF) + w * (16 * H_PITCH);
  const f16* hrow = hscr + lanelo * H_PITCH;

  const float bfc = (lanelo < D_OUT) ? b_fc[lanelo] : 0.0f;

  // initial h from hc1 (f32 -> f16 A-operand regs)
  v16h hA[8];
  {
    const float* hp = hc1 + (size_t)rowg * D_H;
#pragma unroll
    for (int kt = 0; kt < 8; kt++) hA[kt] = load_A_gf32(hp, kt, hsel);
  }

  for (int t = 0; t < T_STEPS; t++) {
    const float* xp = x + ((size_t)t * BATCH + rowg) * D_IN;
    v16h xA[2];
    xA[0] = load_A_gf32(xp, 0, hsel);
    xA[1] = load_A_gf32(xp, 1, hsel);
    if (t < T_STEPS - 1)
      __builtin_prefetch(x + ((size_t)(t + 1) * BATCH + rowg) * D_IN, 0, 1);

    // h_new = tanh(x @ Wi2h^T + h @ Wh2h^T + bsum), 16 N-tiles of 16
    // B-tiles manually software-pipelined through 3 rotating register buffers.
#pragma unroll 2
    for (int nt = 0; nt < 16; nt++) {
      const float bs = s_bsum[nt * 16 + lanelo];
      v8f acc = {0.f, 0.f, 0.f, 0.f, 0.f, 0.f, 0.f, 0.f};
      v16h b0 = load_B_lds(sWi2h, nt * 2 + 0, lane);
      v16h b1 = load_B_lds(sWi2h, nt * 2 + 1, lane);
      v16h b2 = load_B_lds(sWh2h, nt * 8 + 0, lane);
      acc = WMMA(xA[0], b0, acc);
      b0 = load_B_lds(sWh2h, nt * 8 + 1, lane);
      acc = WMMA(xA[1], b1, acc);
      b1 = load_B_lds(sWh2h, nt * 8 + 2, lane);
      acc = WMMA(hA[0], b2, acc);
      b2 = load_B_lds(sWh2h, nt * 8 + 3, lane);
      acc = WMMA(hA[1], b0, acc);
      b0 = load_B_lds(sWh2h, nt * 8 + 4, lane);
      acc = WMMA(hA[2], b1, acc);
      b1 = load_B_lds(sWh2h, nt * 8 + 5, lane);
      acc = WMMA(hA[3], b2, acc);
      b2 = load_B_lds(sWh2h, nt * 8 + 6, lane);
      acc = WMMA(hA[4], b0, acc);
      b0 = load_B_lds(sWh2h, nt * 8 + 7, lane);
      acc = WMMA(hA[5], b1, acc);
      acc = WMMA(hA[6], b2, acc);
      acc = WMMA(hA[7], b0, acc);

      float hv[8];
#pragma unroll
      for (int i = 0; i < 8; i++) hv[i] = fast_tanh(acc[i] + bs);
#pragma unroll
      for (int i = 0; i < 8; i++)
        hscr[(i + 8 * hsel) * H_PITCH + nt * 16 + lanelo] = (f16)hv[i];
    }

    // reload h in A-operand layout (kept in regs for next step and h2o now)
#pragma unroll
    for (int kt = 0; kt < 8; kt++) hA[kt] = load_A_lds(hrow, kt, hsel);

    // mid = tanh(h @ Wh2o^T + b_h2o), 4 N-tiles; scratch reused (h already in regs)
#pragma unroll 2
    for (int nt = 0; nt < 4; nt++) {
      const float bo = s_bh2o[nt * 16 + lanelo];
      v8f acc = {0.f, 0.f, 0.f, 0.f, 0.f, 0.f, 0.f, 0.f};
      v16h b0 = load_B_lds(sWh2o, nt * 8 + 0, lane);
      v16h b1 = load_B_lds(sWh2o, nt * 8 + 1, lane);
      v16h b2 = load_B_lds(sWh2o, nt * 8 + 2, lane);
      acc = WMMA(hA[0], b0, acc);
      b0 = load_B_lds(sWh2o, nt * 8 + 3, lane);
      acc = WMMA(hA[1], b1, acc);
      b1 = load_B_lds(sWh2o, nt * 8 + 4, lane);
      acc = WMMA(hA[2], b2, acc);
      b2 = load_B_lds(sWh2o, nt * 8 + 5, lane);
      acc = WMMA(hA[3], b0, acc);
      b0 = load_B_lds(sWh2o, nt * 8 + 6, lane);
      acc = WMMA(hA[4], b1, acc);
      b1 = load_B_lds(sWh2o, nt * 8 + 7, lane);
      acc = WMMA(hA[5], b2, acc);
      acc = WMMA(hA[6], b0, acc);
      acc = WMMA(hA[7], b1, acc);

      float mv[8];
#pragma unroll
      for (int i = 0; i < 8; i++) mv[i] = fast_tanh(acc[i] + bo);
#pragma unroll
      for (int i = 0; i < 8; i++)
        hscr[(i + 8 * hsel) * H_PITCH + nt * 16 + lanelo] = (f16)mv[i];
    }

    // out = mid @ Wfc^T + b_fc  (N padded 4 -> 16)
    v16h mA0 = load_A_lds(hrow, 0, hsel);
    v16h mA1 = load_A_lds(hrow, 1, hsel);
    v16h bf0 = load_B_lds(sWfc, 0, lane);
    v16h bf1 = load_B_lds(sWfc, 1, lane);
    v8f acc = {0.f, 0.f, 0.f, 0.f, 0.f, 0.f, 0.f, 0.f};
    acc = WMMA(mA0, bf0, acc);
    acc = WMMA(mA1, bf1, acc);
    if (lanelo < D_OUT) {
      float* op = out_seq + ((size_t)t * BATCH + brow0) * D_OUT;
#pragma unroll
      for (int i = 0; i < 8; i++)
        op[(i + 8 * hsel) * D_OUT + lanelo] = acc[i] + bfc;
    }
  }

  // branch-free epilogue: h_final from hA registers (A-operand layout -> f32 rows)
  {
    float* hf = h_final + (size_t)rowg * D_H;
#pragma unroll
    for (int kt = 0; kt < 8; kt++) {
      float* p = hf + kt * 32 + hsel * 8;
      v4f c0, c1, c2, c3;
#pragma unroll
      for (int i = 0; i < 4; i++) {
        c0[i] = (float)hA[kt][i];
        c1[i] = (float)hA[kt][i + 4];
        c2[i] = (float)hA[kt][i + 8];
        c3[i] = (float)hA[kt][i + 12];
      }
      *(v4f*)p        = c0;
      *(v4f*)(p + 4)  = c1;
      *(v4f*)(p + 16) = c2;
      *(v4f*)(p + 20) = c3;
    }
  }
}

extern "C" void kernel_launch(void* const* d_in, const int* in_sizes, int n_in,
                              void* d_out, int out_size, void* d_ws, size_t ws_size,
                              hipStream_t stream) {
  (void)in_sizes; (void)n_in; (void)out_size; (void)ws_size;
  const float* x     = (const float*)d_in[0];
  const float* hc1   = (const float*)d_in[1];
  const float* Wi2h  = (const float*)d_in[2];
  const float* b_i2h = (const float*)d_in[3];
  const float* Wh2h  = (const float*)d_in[4];
  const float* b_h2h = (const float*)d_in[5];
  const float* Wh2o  = (const float*)d_in[6];
  const float* b_h2o = (const float*)d_in[7];
  const float* Wfc   = (const float*)d_in[8];
  const float* b_fc  = (const float*)d_in[9];
  f16* ws = (f16*)d_ws;  // needs PK_TOTAL*2 = 198656 bytes of scratch
  float* out_seq = (float*)d_out;
  float* h_final = out_seq + (size_t)T_STEPS * BATCH * D_OUT;

  pack_weights<<<(PK_TOTAL + 255) / 256, 256, 0, stream>>>(Wi2h, Wh2h, Wh2o, Wfc, ws);

  hipFuncSetAttribute((const void*)rnn_wmma_kernel,
                      hipFuncAttributeMaxDynamicSharedMemorySize, SM_TOTAL);
  rnn_wmma_kernel<<<BATCH / (NWAVES * 16), NWAVES * 32, SM_TOTAL, stream>>>(
      x, hc1, b_i2h, b_h2h, b_h2o, b_fc, ws, out_seq, h_final);
}
